// ChildSumTreeLSTMEncoder_87686052315705
// MI455X (gfx1250) — compile-verified, matrix operand order checked by reading, not run on hardware
//
#include <hip/hip_runtime.h>
#include <hip/hip_bf16.h>
#include <stdint.h>

// ---------------------------------------------------------------------------
// Child-Sum TreeLSTM for MI455X (gfx1250, wave32, WMMA)
// N=65536 nodes, E=262144 edges, EDIM=300, HDIM=256
// ---------------------------------------------------------------------------

#define NNODES 65536
#define NEDGES 262144
#define EDIMC  300
#define HDIMC  256

typedef __bf16 v16bf __attribute__((ext_vector_type(16)));
typedef __bf16 v2bf  __attribute__((ext_vector_type(2)));
typedef float  v8f   __attribute__((ext_vector_type(8)));

// pack two f32 -> packed bf16 dword (RNE)
__device__ __forceinline__ unsigned pack_bf16(float a, float b) {
#if __has_builtin(__builtin_amdgcn_cvt_pk_bf16_f32)
    v2bf r = __builtin_amdgcn_cvt_pk_bf16_f32(a, b);
    return __builtin_bit_cast(unsigned, r);
#else
    unsigned ua = __float_as_uint(a);
    unsigned ub = __float_as_uint(b);
    ua += 0x7FFFu + ((ua >> 16) & 1u);
    ub += 0x7FFFu + ((ub >> 16) & 1u);
    return (ua >> 16) | (ub & 0xFFFF0000u);
#endif
}
__device__ __forceinline__ unsigned short f2bf(float f) {
    unsigned int u = __float_as_uint(f);
    u += 0x7FFFu + ((u >> 16) & 1u);
    return (unsigned short)(u >> 16);
}
__device__ __forceinline__ float sigmoidf_(float x) {
    return 1.0f / (1.0f + __expf(-x));
}

// ---------------------------------------------------------------------------
// Weight prep (tiny): transpose + convert to bf16 [Nout][Kpad]
// ---------------------------------------------------------------------------

// Wt_all [1024][576]:
//   n<768 : k<556 -> W_combined[k][n]      (x rows 0..299 + h rows 300..555)
//   n>=768: k<300 -> W_f[k][n-768]
//   K tail zero-padded to 576
__global__ void prep_wall_kernel(const float* __restrict__ Wc,
                                 const float* __restrict__ Wf,
                                 unsigned short* __restrict__ out) {
    int idx = blockIdx.x * 256 + threadIdx.x;          // n*576 + k
    if (idx >= 1024 * 576) return;
    int n = idx / 576, k = idx % 576;
    float v = 0.0f;
    if (n < 768) { if (k < 556) v = Wc[(long)k * 768 + n]; }
    else         { if (k < 300) v = Wf[(long)k * 256 + (n - 768)]; }
    out[idx] = f2bf(v);
}

// Ut [256][256] = U_f^T
__global__ void prep_ut_kernel(const float* __restrict__ Uf,
                               unsigned short* __restrict__ out) {
    int idx = blockIdx.x * 256 + threadIdx.x;          // n*256 + k
    if (idx >= 256 * 256) return;
    int n = idx / 256, k = idx % 256;
    out[idx] = f2bf(Uf[(long)k * 256 + n]);
}

// bias_all [1024] = [b_combined(768) | b_f(256)]  (b_f folded into f_inputs)
__global__ void prep_bias_kernel(const float* __restrict__ bc,
                                 const float* __restrict__ bf,
                                 float* __restrict__ out) {
    int i = blockIdx.x * 256 + threadIdx.x;
    if (i >= 1024) return;
    out[i] = (i < 768) ? bc[i] : bf[i - 768];
}

// ---------------------------------------------------------------------------
// segment_ids is sorted: off[j] = lower_bound(seg, j); off[N] = E falls out
// ---------------------------------------------------------------------------
__global__ void seg_offsets_kernel(const int* __restrict__ seg,
                                   int* __restrict__ off, int E, int N) {
    int j = blockIdx.x * blockDim.x + threadIdx.x;
    if (j > N) return;
    int lo = 0, hi = E;
    while (lo < hi) {
        int mid = (lo + hi) >> 1;
        if (seg[mid] < j) lo = mid + 1; else hi = mid;
    }
    off[j] = lo;
}

// h_tilde[j] = sum of prev_h rows for children of node j (atomic-free)
__global__ __launch_bounds__(256)
void h_tilde_kernel(const float* __restrict__ prev_h,
                    const int* __restrict__ off,
                    float* __restrict__ htil) {
    int j = blockIdx.x, t = threadIdx.x;
    int beg = off[j], end = off[j + 1];
    float s = 0.0f;
    for (int e = beg; e < end; ++e) s += prev_h[(long)e * HDIMC + t];
    htil[(long)j * HDIMC + t] = s;
}

// ---------------------------------------------------------------------------
// WMMA bf16 GEMM with split-K A source (virtual concat [A | A2]) and
// double-buffered LDS pipeline.
//   C[M][ldc] = concat(A,A2)[M][K] x Bt[Nout][ldb]^T (+bias)
//   A covers k in [0,K1), A2 covers k in [K1,K), zero for k in [K,ldb).
// Block: 256 threads / 8 waves; tile 128x128; wave tile 32x64 (2x4 wmma).
// Pipeline: A register-staged one step ahead (HBM stream); B copied
// global->LDS one buffer ahead in the issue phase (short live range, no
// spill). One barrier per K-step.
// ---------------------------------------------------------------------------
__global__ __launch_bounds__(256, 2)
void wmma_gemm_bf16_kernel(const float* __restrict__ A,  int lda,
                           const float* __restrict__ A2, int lda2,
                           int K1, int K,
                           const unsigned short* __restrict__ Bt, int ldb,
                           float* __restrict__ C, int ldc,
                           const float* __restrict__ bias) {
    __shared__ __align__(16) unsigned short sA[2][128 * 40];  // +8 hw pad/row
    __shared__ __align__(16) unsigned short sB[2][128 * 40];

    const int tid  = threadIdx.x;
    const int lane = tid & 31;
    const int wave = tid >> 5;
    const int wm   = wave & 3;        // 0..3 -> M sub-tile *32
    const int wn   = wave >> 2;       // 0..1 -> N sub-tile *64
    const int half = lane >> 4;       // K-half per ISA fragment layout
    const int r16  = lane & 15;       // row (A) / col (B/C) within 16
    const long blockM = (long)blockIdx.y * 128;
    const int  blockN = blockIdx.x * 128;

    // ---- accumulators ----
    v8f acc[2][4];
#pragma unroll
    for (int mi = 0; mi < 2; ++mi)
#pragma unroll
        for (int ni = 0; ni < 4; ++ni) {
            int gn = blockN + wn * 64 + ni * 16 + r16;
            float init = bias ? bias[gn] : 0.0f;
#pragma unroll
            for (int r = 0; r < 8; ++r) acc[mi][ni][r] = init;
        }

    // ---- A tile: global -> regs (held across the WMMA phase) ----
    float4 ra[4];
    auto loadA = [&](int k0) {
#pragma unroll
        for (int i = 0; i < 4; ++i) {
            int idx = tid + i * 256;              // 1024 float4 chunks
            int row = idx >> 3;
            int c4  = idx & 7;
            int kg  = k0 + c4 * 4;
            float4 v = make_float4(0.f, 0.f, 0.f, 0.f);
            if (kg < K1)
                v = *reinterpret_cast<const float4*>(
                        A + (blockM + row) * (long)lda + kg);
            else if (kg < K)
                v = *reinterpret_cast<const float4*>(
                        A2 + (blockM + row) * (long)lda2 + (kg - K1));
            ra[i] = v;
        }
    };
    // ---- B tile: global -> LDS directly (short live range) ----
    auto copyB = [&](int k0, int buf) {
#pragma unroll
        for (int i = 0; i < 2; ++i) {
            int idx = tid + i * 256;              // 512 uint4 chunks
            int n  = idx >> 2;
            int c8 = idx & 3;
            uint4 q = *reinterpret_cast<const uint4*>(
                          Bt + (long)(blockN + n) * ldb + k0 + c8 * 8);
            reinterpret_cast<uint4*>(sB[buf])[n * 5 + c8] = q;
        }
    };

    const int steps = ldb / 32;   // ldb == Kpad
    int cur = 0;
    copyB(0, 0);
    loadA(0);

    union Frag { uint4 q[2]; v16bf v; };

    for (int s = 0; s < steps; ++s) {
        // ---- commit staged A regs to LDS buffer `cur` (f32 -> bf16) ----
#pragma unroll
        for (int i = 0; i < 4; ++i) {
            int idx = tid + i * 256;
            int row = idx >> 3;
            int c4  = idx & 7;
            uint2 p;
            p.x = pack_bf16(ra[i].x, ra[i].y);
            p.y = pack_bf16(ra[i].z, ra[i].w);
            reinterpret_cast<uint2*>(sA[cur])[row * 10 + c4] = p;
        }
        __syncthreads();

        // ---- issue phase: next B tile -> sB[cur^1]; next A loads in
        //      flight through the WMMAs (B issued+waited first, A after) ----
        if (s + 1 < steps) {
            copyB((s + 1) * 32, cur ^ 1);
            loadA((s + 1) * 32);
        }

        // ---- fragments per ISA 16-bit 16x32 layout:
        // lane r16    : row r16, K {0..7, 16..23}
        // lane r16+16 : row r16, K {8..15, 24..31}
        const uint4* sA4 = reinterpret_cast<const uint4*>(sA[cur]);
        const uint4* sB4 = reinterpret_cast<const uint4*>(sB[cur]);
        Frag a[2];
#pragma unroll
        for (int mi = 0; mi < 2; ++mi) {
            int base = (wm * 32 + mi * 16 + r16) * 5 + half;  // uint4 units
            a[mi].q[0] = sA4[base];
            a[mi].q[1] = sA4[base + 2];
        }
        // stream B fragments one at a time (small live range)
#pragma unroll
        for (int ni = 0; ni < 4; ++ni) {
            Frag b;
            int base = (wn * 64 + ni * 16 + r16) * 5 + half;
            b.q[0] = sB4[base];
            b.q[1] = sB4[base + 2];
#pragma unroll
            for (int mi = 0; mi < 2; ++mi)
                acc[mi][ni] = __builtin_amdgcn_wmma_f32_16x16x32_bf16(
                    false, a[mi].v, false, b.v,
                    (short)0, acc[mi][ni], false, false);
        }

        cur ^= 1;
    }

    // ---- store C (ISA 16x16 f32 C/D layout) ----
#pragma unroll
    for (int mi = 0; mi < 2; ++mi)
#pragma unroll
        for (int ni = 0; ni < 4; ++ni) {
            int gn = blockN + wn * 64 + ni * 16 + r16;
            long gm0 = blockM + wm * 32 + mi * 16 + half * 8;
#pragma unroll
            for (int r = 0; r < 8; ++r)
                C[(gm0 + r) * ldc + gn] = acc[mi][ni][r];
        }
}

// ---------------------------------------------------------------------------
// Per-node fused edge-gate reduction + final gates. One block per node.
// big[j] = [z_i(256) | z_o(256) | z_u(256) | f_in+b_f(256)]
// ---------------------------------------------------------------------------
__global__ __launch_bounds__(256)
void edge_final_kernel(const float* __restrict__ big,
                       const float* __restrict__ hU,
                       const float* __restrict__ prev_c,
                       const int* __restrict__ off,
                       float* __restrict__ out) {
    int j = blockIdx.x, t = threadIdx.x;
    long jb = (long)j * 1024;
    float fin = big[jb + 768 + t];           // (x@W_f)[j] + b_f
    int beg = off[j], end = off[j + 1];
    float fc = 0.0f;
    for (int e = beg; e < end; ++e) {
        float f = sigmoidf_(fin + hU[(long)e * HDIMC + t]);
        fc += f * prev_c[(long)e * HDIMC + t];
    }
    float zi = big[jb + t];
    float zo = big[jb + 256 + t];
    float zu = big[jb + 512 + t];
    float c = sigmoidf_(zi) * tanhf(zu) + fc;
    float h = sigmoidf_(zo) * tanhf(c);
    out[(long)j * HDIMC + t] = c;
    out[(long)NNODES * HDIMC + (long)j * HDIMC + t] = h;
}

// ---------------------------------------------------------------------------
extern "C" void kernel_launch(void* const* d_in, const int* in_sizes, int n_in,
                              void* d_out, int out_size, void* d_ws, size_t ws_size,
                              hipStream_t stream) {
    (void)in_sizes; (void)n_in; (void)out_size; (void)ws_size;

    const float* x   = (const float*)d_in[0];   // [N,300]
    const float* pc  = (const float*)d_in[1];   // [E,256]
    const float* ph  = (const float*)d_in[2];   // [E,256]
    const float* Wc  = (const float*)d_in[3];   // [556,768]
    const float* bc  = (const float*)d_in[4];   // [768]
    const float* Wf  = (const float*)d_in[5];   // [300,256]
    const float* Uf  = (const float*)d_in[6];   // [256,256]
    const float* bf  = (const float*)d_in[7];   // [256]
    const int*   seg = (const int*)d_in[8];     // [E], sorted

    // workspace layout (all chunks 16B aligned)
    char* ws = (char*)d_ws;
    size_t o = 0;
    float* big  = (float*)(ws + o); o += (size_t)NNODES * 1024 * 4;  // 268 MB
    float* hU   = (float*)(ws + o); o += (size_t)NEDGES * 256 * 4;   // 268 MB
    float* htil = (float*)(ws + o); o += (size_t)NNODES * 256 * 4;   //  67 MB
    unsigned short* wall = (unsigned short*)(ws + o); o += (size_t)1024 * 576 * 2;
    unsigned short* ut   = (unsigned short*)(ws + o); o += 256 * 256 * 2;
    float* bias = (float*)(ws + o); o += 1024 * 4;
    int*   offs = (int*)(ws + o);   o += (size_t)(NNODES + 2) * 4;

    // 1) weight prep + segment offsets (independent)
    prep_wall_kernel<<<(1024 * 576 + 255) / 256, 256, 0, stream>>>(Wc, Wf, wall);
    prep_ut_kernel  <<<(256 * 256 + 255) / 256, 256, 0, stream>>>(Uf, ut);
    prep_bias_kernel<<<4, 256, 0, stream>>>(bc, bf, bias);
    seg_offsets_kernel<<<(NNODES + 1 + 255) / 256, 256, 0, stream>>>(
        seg, offs, NEDGES, NNODES);

    // 2) h_tilde = segment_sum(prev_h)
    h_tilde_kernel<<<NNODES, 256, 0, stream>>>(ph, offs, htil);

    // 3) big = [x | h_tilde] @ [W_combined | W_f-padded] + [b_combined | b_f]
    //    single fused GEMM, K = 556 padded to 576
    dim3 g1(1024 / 128, NNODES / 128);
    wmma_gemm_bf16_kernel<<<g1, 256, 0, stream>>>(
        x, EDIMC, htil, HDIMC, EDIMC, 556, wall, 576, big, 1024, bias);

    // 4) hU = prev_h @ U_f
    dim3 g3(256 / 128, NEDGES / 128);
    wmma_gemm_bf16_kernel<<<g3, 256, 0, stream>>>(
        ph, HDIMC, ph, HDIMC, HDIMC, HDIMC, ut, HDIMC, hU, HDIMC, nullptr);

    // 5) fused edge gates + segment_sum + final c,h
    edge_final_kernel<<<NNODES, 256, 0, stream>>>(big, hU, pc, offs,
                                                  (float*)d_out);
}
